// MoDBlock_39316130627986
// MI455X (gfx1250) — compile-verified, matrix operand order checked by reading, not run on hardware
//
#include <hip/hip_runtime.h>
#include <hip/hip_bf16.h>
#include <cstdint>

typedef __attribute__((ext_vector_type(16))) __bf16 v16bf;
typedef __attribute__((ext_vector_type(8)))  __bf16 v8bf;
typedef __attribute__((ext_vector_type(4)))  __bf16 v4bf;
typedef __attribute__((ext_vector_type(8)))  float  v8f;

static constexpr int B_ = 4, L_ = 4096, D_ = 2048, H_ = 8192;
static constexpr int CAP = 2048;            // L * CAPACITY_FACTOR
static constexpr int MTOK = B_ * CAP;       // 8192 selected tokens
static constexpr float AUX_W = 0.01f, CAPF = 0.5f;

// gelu(v) = 0.5*v*(1+tanh(k0*(v+k1 v^3))) = v * sigmoid(2*k0*(v+k1 v^3)); branch-free
__device__ __forceinline__ float gelu_fast(float v) {
  const float k0 = 0.7978845608028654f, k1 = 0.044715f;
  float u = k0 * (v + k1 * v * v * v);
  return v / (1.0f + __expf(-2.0f * u));
}

__device__ __forceinline__ v8bf cvt8(const float4& a, const float4& b) {
  v8bf o;
  o[0] = (__bf16)a.x; o[1] = (__bf16)a.y; o[2] = (__bf16)a.z; o[3] = (__bf16)a.w;
  o[4] = (__bf16)b.x; o[5] = (__bf16)b.y; o[6] = (__bf16)b.z; o[7] = (__bf16)b.w;
  return o;
}

// ---------------- router: logits + sigmoid probs (one wave per token) ----------
__global__ __launch_bounds__(256) void router_kernel(
    const float* __restrict__ x, const float* __restrict__ wr,
    float* __restrict__ logits, float* __restrict__ probs) {
  int wid = threadIdx.x >> 5, lane = threadIdx.x & 31;
  int token = blockIdx.x * 8 + wid;                  // B_*L_ tokens total
  const float* xr = x + (size_t)token * D_;
  float s = 0.f;
#pragma unroll
  for (int i = 0; i < 16; ++i) {                     // 16 * 32 lanes * 4 = 2048
    int idx = (i * 32 + lane) * 4;
    float4 v = *(const float4*)(xr + idx);
    float4 w = *(const float4*)(wr + idx);
    s += v.x * w.x + v.y * w.y + v.z * w.z + v.w * w.w;
  }
#pragma unroll
  for (int off = 16; off >= 1; off >>= 1) s += __shfl_xor(s, off, 32);
  if (lane == 0) {
    logits[token] = s;
    probs[token]  = 1.0f / (1.0f + expf(-s));
  }
}

// ---------------- exact top-k via ranking (cap = L/2, ties -> lower index) -----
__global__ __launch_bounds__(256) void topk_rank_kernel(
    const float* __restrict__ logits, int* __restrict__ slot,
    int* __restrict__ idx_list) {
  __shared__ float Ls[L_];                           // 16 KB
  int b   = blockIdx.x / (L_ / 256);
  int blk = blockIdx.x % (L_ / 256);
  const float* lrow = logits + (size_t)b * L_;
  for (int i = threadIdx.x; i < L_; i += 256) Ls[i] = lrow[i];
  __syncthreads();
  int l = blk * 256 + threadIdx.x;
  float mv = Ls[l];
  int rank = 0;
  for (int j = 0; j < L_; ++j) {
    float v = Ls[j];
    rank += (v > mv) || (v == mv && j < l);
  }
  if (rank < CAP) { slot[b * L_ + l] = rank; idx_list[b * CAP + rank] = l; }
  else            { slot[b * L_ + l] = -1; }
}

// ------- fused fp32->bf16 convert + transpose: out[c][r] = bf16(in[r][c]) ------
// in: R x C fp32, out: C x R bf16. 32x32 tiles via LDS.
__global__ __launch_bounds__(256) void cvtT_bf16_kernel(
    const float* __restrict__ in, __bf16* __restrict__ out, int R, int C) {
  __shared__ float T[32][33];
  int tiles_c = C / 32;
  int r0 = (blockIdx.x / tiles_c) * 32;
  int c0 = (blockIdx.x % tiles_c) * 32;
  int r  = threadIdx.x >> 3;           // 0..31
  int c4 = (threadIdx.x & 7) * 4;      // 0,4,..,28
  float4 v = *(const float4*)(in + (size_t)(r0 + r) * C + c0 + c4);
  T[c4 + 0][r] = v.x; T[c4 + 1][r] = v.y; T[c4 + 2][r] = v.z; T[c4 + 3][r] = v.w;
  __syncthreads();
  v4bf o;
  o[0] = (__bf16)T[r][c4 + 0]; o[1] = (__bf16)T[r][c4 + 1];
  o[2] = (__bf16)T[r][c4 + 2]; o[3] = (__bf16)T[r][c4 + 3];
  *(v4bf*)(out + (size_t)(c0 + r) * R + r0 + c4) = o;
}

// ---------------- gather selected tokens (compact, bf16) -----------------------
__global__ __launch_bounds__(256) void gather_kernel(
    const float* __restrict__ x, const int* __restrict__ idx_list,
    __bf16* __restrict__ Xs) {
  int b = blockIdx.x / CAP, r = blockIdx.x % CAP;
  int idx = idx_list[b * CAP + r];
  const float* src = x + ((size_t)b * L_ + idx) * D_;
  __bf16* dst = Xs + ((size_t)b * CAP + r) * D_;
  int base = threadIdx.x * 8;
  float4 a = *(const float4*)(src + base);
  float4 c = *(const float4*)(src + base + 4);
  *(v8bf*)(dst + base) = cvt8(a, c);
}

// ---------------- WMMA bf16 GEMM: C = epi(A @ BT^T + bias) ---------------------
// A: M x K row-major bf16.  BT: N x K row-major bf16 (pre-transposed weight).
// 128x128 block tile, 8 waves (4x2), wave tile 32x64 = 2x4 v_wmma_f32_16x16x32_bf16.
// BK=32; both tiles staged in LDS with 16B vector stores; register-prefetch
// double buffering overlaps next tile's global loads with WMMA compute.
template <int EPI>  // 0: gelu -> bf16 out, 1: +bias -> f32 out
__global__ __launch_bounds__(256) void gemm_kernel(
    const __bf16* __restrict__ A, const __bf16* __restrict__ BT,
    const float* __restrict__ bias, __bf16* __restrict__ outB,
    float* __restrict__ outF, int M, int N, int K) {
  constexpr int PAD = 40;                            // halves; 80B row stride (16B-aligned)
  __shared__ __bf16 As[128 * PAD];
  __shared__ __bf16 Bs[128 * PAD];                   // [n][k], K-contiguous
  int tid = threadIdx.x;
  int wid = tid >> 5, lane = tid & 31;
  int wm = wid >> 1, wn = wid & 1;                   // wave grid 4(M) x 2(N)
  int lrow = lane & 15, kg = lane >> 4;              // kgroup split per ISA layout
  int m0 = blockIdx.y * 128, n0 = blockIdx.x * 128;

  // staging assignment: each thread owns 16 halves of one row of each tile
  int srow = tid >> 1, scc = (tid & 1) * 16;
  const __bf16* ga = A  + (size_t)(m0 + srow) * K + scc;
  const __bf16* gb = BT + (size_t)(n0 + srow) * K + scc;

  v8f acc[2][4] = {};

  // prologue: prefetch k0 = 0 tiles into registers
  v8bf ra0 = *(const v8bf*)ga;
  v8bf ra1 = *(const v8bf*)(ga + 8);
  v8bf rb0 = *(const v8bf*)gb;
  v8bf rb1 = *(const v8bf*)(gb + 8);

  for (int k0 = 0; k0 < K; k0 += 32) {
    *(v8bf*)&As[srow * PAD + scc]     = ra0;
    *(v8bf*)&As[srow * PAD + scc + 8] = ra1;
    *(v8bf*)&Bs[srow * PAD + scc]     = rb0;
    *(v8bf*)&Bs[srow * PAD + scc + 8] = rb1;
    __syncthreads();

    if (k0 + 32 < K) {                               // prefetch next K-tile
      ra0 = *(const v8bf*)(ga + k0 + 32);
      ra1 = *(const v8bf*)(ga + k0 + 40);
      rb0 = *(const v8bf*)(gb + k0 + 32);
      rb1 = *(const v8bf*)(gb + k0 + 40);
    }

    v16bf af[2], bf[4];
#pragma unroll
    for (int tm = 0; tm < 2; ++tm) {
      int row = wm * 32 + tm * 16 + lrow;
      v8bf lo = *(const v8bf*)&As[row * PAD + kg * 8];       // K = kg*8 .. +7
      v8bf hi = *(const v8bf*)&As[row * PAD + kg * 8 + 16];  // K = 16+kg*8 .. +7
      af[tm] = __builtin_shufflevector(lo, hi, 0,1,2,3,4,5,6,7,8,9,10,11,12,13,14,15);
    }
#pragma unroll
    for (int tn = 0; tn < 4; ++tn) {
      int col = wn * 64 + tn * 16 + lrow;
      v8bf lo = *(const v8bf*)&Bs[col * PAD + kg * 8];
      v8bf hi = *(const v8bf*)&Bs[col * PAD + kg * 8 + 16];
      bf[tn] = __builtin_shufflevector(lo, hi, 0,1,2,3,4,5,6,7,8,9,10,11,12,13,14,15);
    }
#pragma unroll
    for (int tm = 0; tm < 2; ++tm)
#pragma unroll
      for (int tn = 0; tn < 4; ++tn)
        acc[tm][tn] = __builtin_amdgcn_wmma_f32_16x16x32_bf16(
            false, af[tm], false, bf[tn], (short)0, acc[tm][tn], false, false);
    __syncthreads();
  }

  // epilogue: C layout VGPR r, lanes 0-15: (M=r, N=lane); 16-31: (M=r+8, N=lane-16)
#pragma unroll
  for (int tm = 0; tm < 2; ++tm) {
#pragma unroll
    for (int tn = 0; tn < 4; ++tn) {
      int col = n0 + wn * 64 + tn * 16 + lrow;
      float bv = bias[col];
#pragma unroll
      for (int r = 0; r < 8; ++r) {
        int row = m0 + wm * 32 + tm * 16 + r + 8 * kg;
        float v = acc[tm][tn][r] + bv;
        if (EPI == 0) outB[(size_t)row * N + col] = (__bf16)gelu_fast(v);
        else          outF[(size_t)row * N + col] = v;
      }
    }
  }
}

// ---------------- blend: selected -> w*ffn + (1-w)*x, else x exactly -----------
__global__ __launch_bounds__(256) void blend_kernel(
    const float* __restrict__ x, const float* __restrict__ probs,
    const int* __restrict__ slot, const float* __restrict__ Y,
    float* __restrict__ out) {
  size_t id = (size_t)blockIdx.x * 256 + threadIdx.x;  // one float4 per thread
  int token = (int)(id / (D_ / 4));
  int d = (int)(id % (D_ / 4)) * 4;
  int b = token / L_;
  float4 xv = *(const float4*)(x + (size_t)token * D_ + d);
  float4 o = xv;
  int s = slot[token];
  if (s >= 0) {
    float w = probs[token];
    float4 y = *(const float4*)(Y + ((size_t)b * CAP + s) * D_ + d);
    o.x = w * y.x + (1.0f - w) * xv.x;
    o.y = w * y.y + (1.0f - w) * xv.y;
    o.z = w * y.z + (1.0f - w) * xv.z;
    o.w = w * y.w + (1.0f - w) * xv.w;
  }
  *(float4*)(out + (size_t)token * D_ + d) = o;
}

// ---------------- deterministic aux loss reduction -----------------------------
__global__ __launch_bounds__(256) void aux_kernel(
    const float* __restrict__ probs, float* __restrict__ out_aux) {
  __shared__ float red[256];
  float total = 0.f;
  for (int b = 0; b < B_; ++b) {
    float s = 0.f;
    for (int i = threadIdx.x; i < L_; i += 256) s += probs[b * L_ + i];
    red[threadIdx.x] = s;
    __syncthreads();
    for (int off = 128; off >= 1; off >>= 1) {
      if (threadIdx.x < off) red[threadIdx.x] += red[threadIdx.x + off];
      __syncthreads();
    }
    if (threadIdx.x == 0) {
      float m = red[0] / (float)L_;
      total += (m - CAPF) * (m - CAPF);
    }
    __syncthreads();
  }
  if (threadIdx.x == 0) *out_aux = AUX_W * (total / (float)B_);
}

extern "C" void kernel_launch(void* const* d_in, const int* in_sizes, int n_in,
                              void* d_out, int out_size, void* d_ws, size_t ws_size,
                              hipStream_t stream) {
  (void)in_sizes; (void)n_in; (void)out_size; (void)ws_size;
  const float* x  = (const float*)d_in[0];
  const float* wr = (const float*)d_in[1];
  const float* W1 = (const float*)d_in[2];
  const float* b1 = (const float*)d_in[3];
  const float* W2 = (const float*)d_in[4];
  const float* b2 = (const float*)d_in[5];
  float* out = (float*)d_out;

  char* ws = (char*)d_ws;
  auto alloc = [&](size_t bytes) {
    char* p = ws; ws += (bytes + 255) & ~(size_t)255; return p;
  };
  float*  logits = (float*)alloc(sizeof(float) * B_ * L_);
  float*  probs  = (float*)alloc(sizeof(float) * B_ * L_);
  int*    slot   = (int*)  alloc(sizeof(int)   * B_ * L_);
  int*    idxl   = (int*)  alloc(sizeof(int)   * B_ * CAP);
  __bf16* W1T = (__bf16*)alloc(sizeof(__bf16) * (size_t)H_ * D_);  // [H][D]
  __bf16* W2T = (__bf16*)alloc(sizeof(__bf16) * (size_t)D_ * H_);  // [D][H]
  __bf16* Xs  = (__bf16*)alloc(sizeof(__bf16) * (size_t)MTOK * D_);
  __bf16* Hs  = (__bf16*)alloc(sizeof(__bf16) * (size_t)MTOK * H_);
  float*  Y   = (float*) alloc(sizeof(float)  * (size_t)MTOK * D_);

  // weight convert + transpose (once per launch): W1 [D][H] -> W1T [H][D], etc.
  cvtT_bf16_kernel<<<(D_ / 32) * (H_ / 32), 256, 0, stream>>>(W1, W1T, D_, H_);
  cvtT_bf16_kernel<<<(H_ / 32) * (D_ / 32), 256, 0, stream>>>(W2, W2T, H_, D_);

  // router: one wave per token, 8 tokens/block
  router_kernel<<<(B_ * L_) / 8, 256, 0, stream>>>(x, wr, logits, probs);

  // exact top-k by ranking: B * (L/256) blocks
  topk_rank_kernel<<<B_ * (L_ / 256), 256, 0, stream>>>(logits, slot, idxl);

  // gather compact bf16 token matrix
  gather_kernel<<<B_ * CAP, 256, 0, stream>>>(x, idxl, Xs);

  // FFN GEMM1: [8192 x 2048] @ [2048 x 8192], gelu epilogue -> bf16
  gemm_kernel<0><<<dim3(H_ / 128, MTOK / 128), 256, 0, stream>>>(
      Xs, W1T, b1, Hs, nullptr, MTOK, H_, D_);

  // FFN GEMM2: [8192 x 8192] @ [8192 x 2048], bias epilogue -> f32
  gemm_kernel<1><<<dim3(D_ / 128, MTOK / 128), 256, 0, stream>>>(
      Hs, W2T, b2, nullptr, Y, MTOK, D_, H_);

  // sigmoid-blended scatter/residual: B*L*D/4 float4 threads
  blend_kernel<<<(B_ * L_ * (D_ / 4)) / 256, 256, 0, stream>>>(
      x, probs, slot, Y, out);

  // aux loss scalar appended after the [B,L,D] output
  aux_kernel<<<1, 256, 0, stream>>>(probs, out + (size_t)B_ * L_ * D_);
}
